// SwinUNet_66065186947383
// MI455X (gfx1250) — matrix operand order repaired and checked by reading, not compiled, not run on hardware
//
#include <hip/hip_runtime.h>
#include <math.h>

// ---------------------------------------------------------------------------
// Types for CDNA5 WMMA (wave32, V_WMMA_F32_16X16X32_F16)
// ---------------------------------------------------------------------------
typedef _Float16 v16h __attribute__((ext_vector_type(16)));
typedef _Float16 h8   __attribute__((ext_vector_type(8)));
typedef _Float16 h4   __attribute__((ext_vector_type(4)));
typedef float    v8f  __attribute__((ext_vector_type(8)));

__device__ __forceinline__ v16h frag16(const _Float16* p0, const _Float16* p1) {
    // p0/p1 must be 16-byte aligned (LDS strides are multiples of 8 halves)
    h8 lo = *(const h8*)p0;
    h8 hi = *(const h8*)p1;
    return __builtin_shufflevector(lo, hi, 0, 1, 2, 3, 4, 5, 6, 7, 8, 9, 10, 11, 12, 13, 14, 15);
}

__device__ __forceinline__ v8f wmma16(v16h a, v16h b, v8f c) {
    return __builtin_amdgcn_wmma_f32_16x16x32_f16(false, a, false, b, (short)0, c, false, false);
}

// ---------------------------------------------------------------------------
// WMMA GEMM: out[M,N] = epi(Ah[M,K](f16) @ Wt[N,K](f16, pre-transposed)).
// Block = 128 threads = 4 waves; tile 64(M) x 128(N); each wave owns 64x32
// (2 B-frags, 8 accumulators -> 8 WMMA per k-step).
// Requirement (all call sites satisfy it): K % 4 == 0.
// Both operands are k-contiguous in global AND in LDS -> all staging is
// clamped h4 (8B) loads + aligned 8B LDS stores, branch-free.
// act: 0 none, 1 exact GELU. bias/res f32, nullable; res may alias out (f32).
// out_half: 1 -> store f16, 0 -> store f32.
// ---------------------------------------------------------------------------
#define GSTR 40  // LDS row stride in halves (multiple of 8 -> 16B aligned frags)

__global__ __launch_bounds__(128) void k_gemm(
    const _Float16* __restrict__ Ah, const _Float16* __restrict__ Wt,
    const float* bias, const float* res, void* outp,
    int M, int N, int K, int act, int out_half)
{
    __shared__ _Float16 As[64 * GSTR];   // As[m][k], k contiguous
    __shared__ _Float16 Bs[128 * GSTR];  // Bs[n][k], k contiguous
    const int tid  = threadIdx.x;
    const int lane = tid & 31;
    const int wave = tid >> 5;
    const int grp  = lane >> 4;
    const int l16  = lane & 15;
    const int m0 = blockIdx.y * 64;
    const int n0 = blockIdx.x * 128;
    const int nb = wave * 32;

    v8f acc[4][2];
#pragma unroll
    for (int i = 0; i < 4; ++i) { acc[i][0] = v8f{}; acc[i][1] = v8f{}; }

    for (int k0 = 0; k0 < K; k0 += 32) {
        if (k0 + 32 < K) {  // prefetch next k-tile (global_prefetch_b8)
            __builtin_prefetch(&Ah[(size_t)min(m0 + (tid >> 3), M - 1) * K +
                                   min(k0 + 32 + (tid & 7) * 4, K - 4)], 0, 1);
            __builtin_prefetch(&Wt[(size_t)min(n0 + (tid >> 3), N - 1) * K +
                                   min(k0 + 32 + (tid & 7) * 4, K - 4)], 0, 1);
        }
        // A tile: 64 rows x 32 k = 512 h4 chunks; 4 per thread.
        h4 va[4]; bool ia[4];
#pragma unroll
        for (int i = 0; i < 4; ++i) {
            int c = tid + i * 128;
            int m = c >> 3, koff = (c & 7) * 4;
            int gm = m0 + m, gk = k0 + koff;
            ia[i] = (gm < M) && (gk < K);
            size_t e = (size_t)min(gm, M - 1) * K + min(gk, K - 4);
            va[i] = *(const h4*)(Ah + e);
        }
        // W tile: 128 rows (n) x 32 k = 1024 h4 chunks; 8 per thread.
        h4 vb[8]; bool ib[8];
#pragma unroll
        for (int i = 0; i < 8; ++i) {
            int c = tid + i * 128;
            int n = c >> 3, koff = (c & 7) * 4;
            int gn = n0 + n, gk = k0 + koff;
            ib[i] = (gn < N) && (gk < K);
            size_t e = (size_t)min(gn, N - 1) * K + min(gk, K - 4);
            vb[i] = *(const h4*)(Wt + e);
        }
        __syncthreads();  // previous iteration's LDS reads complete (WAR)
        h4 z = {};
#pragma unroll
        for (int i = 0; i < 4; ++i) {
            int c = tid + i * 128;
            int m = c >> 3, koff = (c & 7) * 4;
            *(h4*)&As[m * GSTR + koff] = ia[i] ? va[i] : z;
        }
#pragma unroll
        for (int i = 0; i < 8; ++i) {
            int c = tid + i * 128;
            int n = c >> 3, koff = (c & 7) * 4;
            *(h4*)&Bs[n * GSTR + koff] = ib[i] ? vb[i] : z;
        }
        __syncthreads();

        const _Float16* bp0 = &Bs[(nb + l16) * GSTR];
        v16h bf0 = frag16(bp0 + grp * 8, bp0 + 16 + grp * 8);
        const _Float16* bp1 = &Bs[(nb + 16 + l16) * GSTR];
        v16h bf1 = frag16(bp1 + grp * 8, bp1 + 16 + grp * 8);
#pragma unroll
        for (int mt = 0; mt < 4; ++mt) {
            const _Float16* ap = &As[(mt * 16 + l16) * GSTR];
            v16h af = frag16(ap + grp * 8, ap + 16 + grp * 8);
            acc[mt][0] = wmma16(af, bf0, acc[mt][0]);
            acc[mt][1] = wmma16(af, bf1, acc[mt][1]);
        }
    }

    float* outf = (float*)outp;
    _Float16* outh = (_Float16*)outp;
#pragma unroll
    for (int hh = 0; hh < 2; ++hh) {
        const int n = n0 + nb + hh * 16 + l16;
#pragma unroll
        for (int mt = 0; mt < 4; ++mt) {
#pragma unroll
            for (int v = 0; v < 8; ++v) {
                int m = m0 + mt * 16 + v + grp * 8;
                if (m < M && n < N) {
                    float val = acc[mt][hh][v];
                    if (bias) val += bias[n];
                    if (act == 1) val = 0.5f * val * (1.f + erff(val * 0.70710678118654752f));
                    if (res) val += res[(size_t)m * N + n];
                    if (out_half) outh[(size_t)m * N + n] = (_Float16)val;
                    else          outf[(size_t)m * N + n] = val;
                }
            }
        }
    }
}

// ---------------------------------------------------------------------------
// Fused 7x7 window attention, head-dim 32 (true for every stage).
// One wave per (window, head). qkv f16 [Ntok, 3C] token-major, cols
// [q|k|v], within each: head*32 + d. o: f16 [Ntok, C].
// ---------------------------------------------------------------------------
__global__ __launch_bounds__(32) void k_win_attn(
    const _Float16* __restrict__ qkv, const float* __restrict__ rpb,
    _Float16* __restrict__ o, int H, int Wd, int C, int heads, int nh, int nw)
{
    __shared__ _Float16 Qs[64 * 40];  // Qs[tok][d]
    __shared__ _Float16 Ks[64 * 40];  // Ks[tok][d]
    __shared__ _Float16 Vs[32 * 72];  // Vs[d][tok]
    __shared__ _Float16 At[64 * 72];  // attn[row][key] f16
    __shared__ float    Sc[64 * 66];  // scores f32

    const int lane = threadIdx.x;
    const int grp = lane >> 4, l16 = lane & 15;
    const int head = blockIdx.x % heads;
    const int wid  = blockIdx.x / heads;
    const int wxy  = wid % (nh * nw);
    const int b    = wid / (nh * nw);
    const int wy = wxy / nw, wx = wxy % nw;
    const int C3 = 3 * C;

    for (int i = lane; i < 64 * 40; i += 32) { Qs[i] = (_Float16)0.f; Ks[i] = (_Float16)0.f; }
    for (int i = lane; i < 32 * 72; i += 32) Vs[i] = (_Float16)0.f;
    __syncthreads();

    // Stage Q/K/V: h4 (8-byte) loads, f16 end-to-end, no conversions.
    for (int idx = lane; idx < 49 * 8; idx += 32) {
        int t = idx >> 3, off = (idx & 7) * 4;
        int ty = t / 7, tx = t % 7;
        size_t row = (size_t)((b * H + wy * 7 + ty) * Wd + wx * 7 + tx);
        const _Float16* base = qkv + row * C3 + head * 32 + off;
        h4 q = *(const h4*)base;
        h4 k = *(const h4*)(base + C);
        h4 v = *(const h4*)(base + 2 * C);
        *(h4*)&Qs[t * 40 + off] = q;
        *(h4*)&Ks[t * 40 + off] = k;
        Vs[(off + 0) * 72 + t] = v[0];
        Vs[(off + 1) * 72 + t] = v[1];
        Vs[(off + 2) * 72 + t] = v[2];
        Vs[(off + 3) * 72 + t] = v[3];
    }
    __syncthreads();

    // scores = Q @ K^T (single K-step of 32 = head dim)
#pragma unroll
    for (int mt = 0; mt < 4; ++mt) {
        const _Float16* qr = &Qs[(mt * 16 + l16) * 40];
        v16h af = frag16(qr + grp * 8, qr + 16 + grp * 8);
#pragma unroll
        for (int nt = 0; nt < 4; ++nt) {
            const _Float16* kr = &Ks[(nt * 16 + l16) * 40];
            v16h bf = frag16(kr + grp * 8, kr + 16 + grp * 8);
            v8f c{};
            c = wmma16(af, bf, c);
#pragma unroll
            for (int v = 0; v < 8; ++v)
                Sc[(mt * 16 + v + grp * 8) * 66 + nt * 16 + l16] = c[v];
        }
    }
    __syncthreads();

    // softmax over 49 keys + relative-position bias computed in-kernel
    const float scale = 0.17677669529663687f;  // 1/sqrt(32)
    for (int r = lane; r < 64; r += 32) {
        if (r < 49) {
            int qy = r / 7, qx = r % 7;
            float mx = -1e30f;
            for (int j = 0; j < 49; ++j) {
                int ky = j / 7, kx = j % 7;
                int rel = (qy - ky + 6) * 13 + (qx - kx + 6);
                float v = Sc[r * 66 + j] * scale + rpb[rel * heads + head];
                Sc[r * 66 + j] = v;
                mx = fmaxf(mx, v);
            }
            float sum = 0.f;
            for (int j = 0; j < 49; ++j) {
                float e = __expf(Sc[r * 66 + j] - mx);
                Sc[r * 66 + j] = e;
                sum += e;
            }
            float inv = 1.f / sum;
            for (int j = 0; j < 64; ++j)
                At[r * 72 + j] = (j < 49) ? (_Float16)(Sc[r * 66 + j] * inv) : (_Float16)0.f;
        } else {
            for (int j = 0; j < 64; ++j) At[r * 72 + j] = (_Float16)0.f;
        }
    }
    __syncthreads();

    // out = attn @ V (K = 64 tokens -> 2 k-steps; N = 32 -> 2 n-tiles)
#pragma unroll
    for (int mt = 0; mt < 4; ++mt) {
#pragma unroll
        for (int nt = 0; nt < 2; ++nt) {
            v8f c{};
#pragma unroll
            for (int kt = 0; kt < 2; ++kt) {
                const _Float16* ar = &At[(mt * 16 + l16) * 72 + kt * 32];
                v16h af = frag16(ar + grp * 8, ar + 16 + grp * 8);
                const _Float16* vr = &Vs[(nt * 16 + l16) * 72 + kt * 32];
                v16h bf = frag16(vr + grp * 8, vr + 16 + grp * 8);
                c = wmma16(af, bf, c);
            }
#pragma unroll
            for (int v = 0; v < 8; ++v) {
                int m = mt * 16 + v + grp * 8;
                if (m < 49) {
                    int ty = m / 7, tx = m % 7;
                    size_t row = (size_t)((b * H + wy * 7 + ty) * Wd + wx * 7 + tx);
                    o[row * C + head * 32 + nt * 16 + l16] = (_Float16)c[v];
                }
            }
        }
    }
}

// ---------------------------------------------------------------------------
// LayerNorm over last dim: wave32 per row; OT selects f32 or f16 output.
// ---------------------------------------------------------------------------
template <typename OT>
__global__ void k_layernorm(const float* __restrict__ x, const float* __restrict__ g,
                            const float* __restrict__ bb, OT* __restrict__ out,
                            int M, int C)
{
    int row = blockIdx.x * (blockDim.x >> 5) + (threadIdx.x >> 5);
    if (row >= M) return;
    int lane = threadIdx.x & 31;
    const float* xr = x + (size_t)row * C;
    float s = 0.f, s2 = 0.f;
    for (int c = lane; c < C; c += 32) { float v = xr[c]; s += v; s2 += v * v; }
#pragma unroll
    for (int o = 16; o; o >>= 1) { s += __shfl_xor(s, o, 32); s2 += __shfl_xor(s2, o, 32); }
    float mean = s / (float)C;
    float var  = s2 / (float)C - mean * mean;
    float r = rsqrtf(var + 1e-5f);
    OT* orow = out + (size_t)row * C;
    for (int c = lane; c < C; c += 32) orow[c] = (OT)((xr[c] - mean) * r * g[c] + bb[c]);
}

// ---------------------------------------------------------------------------
// Elementwise / remap kernels
// ---------------------------------------------------------------------------
__global__ void k_cvt(const float* __restrict__ in, _Float16* __restrict__ out, int n)
{
    int idx = blockIdx.x * blockDim.x + threadIdx.x;
    if (idx < n) out[idx] = (_Float16)in[idx];
}

// weight repack: f32 [K,N] -> f16 [N,K] (k-contiguous for GEMM staging)
__global__ void k_cvt_t(const float* __restrict__ in, _Float16* __restrict__ out,
                        int K, int N)
{
    int idx = blockIdx.x * blockDim.x + threadIdx.x;
    if (idx >= K * N) return;
    int n = idx / K, k = idx % K;
    out[idx] = (_Float16)in[(size_t)k * N + n];
}

__global__ void k_im2col(const float* __restrict__ x, _Float16* __restrict__ A)
{
    int idx = blockIdx.x * blockDim.x + threadIdx.x;
    const int total = 4 * 112 * 112 * 48;
    if (idx >= total) return;
    int k = idx % 48, row = idx / 48;
    int w = row % 112, h = (row / 112) % 112, b = row / (112 * 112);
    int q = k % 4, p = (k / 4) % 4, c = k / 16;
    A[idx] = (_Float16)x[(((size_t)b * 3 + c) * 448 + h * 4 + p) * 448 + w * 4 + q];
}

__global__ void k_transpose(const float* __restrict__ in, float* __restrict__ out, int R, int C)
{
    int idx = blockIdx.x * blockDim.x + threadIdx.x;
    if (idx >= R * C) return;
    int r = idx / C, c = idx % C;
    out[(size_t)c * R + r] = in[idx];
}

// final.w (96,48,2,2) -> [K=96][(k*2+l)*48+o]
__global__ void k_repack_final(const float* __restrict__ in, float* __restrict__ out)
{
    int idx = blockIdx.x * blockDim.x + threadIdx.x;
    if (idx >= 96 * 192) return;
    int c = idx / 192, r = idx % 192;
    int kl = r / 48, o = r % 48;
    int k = kl >> 1, l = kl & 1;
    out[idx] = in[(((size_t)c * 48 + o) * 2 + k) * 2 + l];
}

// head.w (19,48) -> [K=48][N=24] zero-padded
__global__ void k_head_repack(const float* __restrict__ in, float* __restrict__ out)
{
    int idx = blockIdx.x * blockDim.x + threadIdx.x;
    if (idx >= 48 * 24) return;
    int k = idx / 24, n = idx % 24;
    out[idx] = (n < 19) ? in[(size_t)n * 48 + k] : 0.f;
}

__global__ void k_head_bias(const float* __restrict__ in, float* __restrict__ out)
{
    int idx = blockIdx.x * blockDim.x + threadIdx.x;
    if (idx < 24) out[idx] = (idx < 19) ? in[idx] : 0.f;
}

// [B,H,W,C] -> [B*(H/2)*(W/2), 4C]  col = (s*2+t)*C + ch, pixel (2i+s,2j+t)
__global__ void k_merge_gather(const float* __restrict__ in, float* __restrict__ out,
                               int Bt, int H, int W, int C)
{
    int idx = blockIdx.x * blockDim.x + threadIdx.x;
    int C4 = 4 * C;
    int total = Bt * (H / 2) * (W / 2) * C4;
    if (idx >= total) return;
    int col = idx % C4, pix = idx / C4;
    int ch = col % C, st = col / C;
    int s = st >> 1, t = st & 1;
    int j = pix % (W / 2), i = (pix / (W / 2)) % (H / 2), b = pix / ((W / 2) * (H / 2));
    out[idx] = in[(((size_t)b * H + 2 * i + s) * W + 2 * j + t) * C + ch];
}

// [B,H,W,4*Co] -> [B,2H,2W,Co]  (patch_expand rearrange)
__global__ void k_expand_rearrange(const float* __restrict__ in, float* __restrict__ out,
                                   int Bt, int H, int W, int Co)
{
    int idx = blockIdx.x * blockDim.x + threadIdx.x;
    int total = Bt * 2 * H * 2 * W * Co;
    if (idx >= total) return;
    int ch = idx % Co;
    int x2 = (idx / Co) % (2 * W);
    int y2 = (idx / (Co * 2 * W)) % (2 * H);
    int b  = idx / (Co * 2 * W * 2 * H);
    int s = y2 & 1, t = x2 & 1, i = y2 >> 1, j = x2 >> 1;
    out[idx] = in[(((size_t)b * H + i) * W + j) * (4 * Co) + (s * 2 + t) * Co + ch];
}

// concat [a | b] along channels, f32 inputs -> f16 output (feeds a GEMM A)
__global__ void k_concat_h(const float* __restrict__ a, const float* __restrict__ bsrc,
                           _Float16* __restrict__ out, int M, int Ca, int Cb)
{
    int idx = blockIdx.x * blockDim.x + threadIdx.x;
    int Ct = Ca + Cb;
    if (idx >= M * Ct) return;
    int c = idx % Ct, m = idx / Ct;
    float v = (c < Ca) ? a[(size_t)m * Ca + c] : bsrc[(size_t)m * Cb + (c - Ca)];
    out[idx] = (_Float16)v;
}

// F[50176,192] (cols = (k,l,o)) -> y[4,224,224,48] + bias[o]
__global__ void k_final_rearrange(const float* __restrict__ F, const float* __restrict__ bias,
                                  float* __restrict__ out)
{
    int idx = blockIdx.x * blockDim.x + threadIdx.x;
    const int total = 4 * 224 * 224 * 48;
    if (idx >= total) return;
    int o  = idx % 48;
    int x2 = (idx / 48) % 224;
    int y2 = (idx / (48 * 224)) % 224;
    int b  = idx / (48 * 224 * 224);
    int k = y2 & 1, h = y2 >> 1, l = x2 & 1, w = x2 >> 1;
    out[idx] = F[(((size_t)b * 112 + h) * 112 + w) * 192 + (k * 2 + l) * 48 + o] + bias[o];
}

// logits [4,224,224,24] (19 valid) -> bilinear -> d_out [4,19,480,480]
__global__ void k_resize(const float* __restrict__ in, float* __restrict__ out)
{
    int idx = blockIdx.x * blockDim.x + threadIdx.x;
    const int total = 4 * 19 * 480 * 480;
    if (idx >= total) return;
    int ox = idx % 480;
    int oy = (idx / 480) % 480;
    int o  = (idx / (480 * 480)) % 19;
    int b  = idx / (480 * 480 * 19);
    const float f = 224.f / 480.f;
    float sy = fminf(fmaxf((oy + 0.5f) * f - 0.5f, 0.f), 223.f);
    float sx = fminf(fmaxf((ox + 0.5f) * f - 0.5f, 0.f), 223.f);
    int y0 = (int)sy, x0 = (int)sx;
    int y1 = min(y0 + 1, 223), x1 = min(x0 + 1, 223);
    float fy = sy - y0, fx = sx - x0;
    const float* p = in + (size_t)b * 224 * 224 * 24 + o;
    auto at = [&](int h, int w) { return p[((size_t)h * 224 + w) * 24]; };
    out[idx] = (1.f - fy) * ((1.f - fx) * at(y0, x0) + fx * at(y0, x1)) +
               fy        * ((1.f - fx) * at(y1, x0) + fx * at(y1, x1));
}

// ---------------------------------------------------------------------------
// Host side. Parameter walk = setup_inputs() insertion order, depth-first:
// x; embed{w,b}; enc_blocks[3]{b1,b2: ln1{g,b},qkv{w,b},proj{w,b},rpb,
// ln2{g,b},fc1{w,b},fc2{w,b}}; enc_merge[3]{ln{g,b},red}; bottleneck;
// dec_expand[3]{exp,ln{g,b}}; dec_skip[3]{w,b}; dec_blocks[3];
// final{w,b,ln{g,b}}; head{w,b}
// ---------------------------------------------------------------------------
namespace {

struct Reader { void* const* d; int i; const float* f() { return (const float*)d[i++]; } };
struct LNW  { const float* g; const float* b; };
struct LinW { const float* w; const float* b; };
struct SwtbW { LNW ln1; LinW qkv; LinW proj; const float* rpb; LNW ln2; LinW fc1; LinW fc2; };
struct BlockW { SwtbW b1, b2; };

LNW  rdLN(Reader& r)  { LNW p; p.g = r.f(); p.b = r.f(); return p; }
LinW rdLin(Reader& r) { LinW p; p.w = r.f(); p.b = r.f(); return p; }
SwtbW rdSwtb(Reader& r) {
    SwtbW s; s.ln1 = rdLN(r); s.qkv = rdLin(r); s.proj = rdLin(r);
    s.rpb = r.f(); s.ln2 = rdLN(r); s.fc1 = rdLin(r); s.fc2 = rdLin(r); return s;
}
BlockW rdBlock(Reader& r) { BlockW b; b.b1 = rdSwtb(r); b.b2 = rdSwtb(r); return b; }

inline int ceil_div(int a, int b) { return (a + b - 1) / b; }

// GEMM wrapper: transpose-converts the f32 [K,N] weight into f16 [N,K]
// scratch HW, then launches the WMMA GEMM.
void gemm(hipStream_t st, const _Float16* Ah, const float* Wf, const float* bias,
          const float* res, void* out, bool out_half, int M, int N, int K, int act,
          _Float16* HW)
{
    int wn = K * N;
    k_cvt_t<<<ceil_div(wn, 256), 256, 0, st>>>(Wf, HW, K, N);
    dim3 g(ceil_div(N, 128), ceil_div(M, 64));
    k_gemm<<<g, dim3(128), 0, st>>>(Ah, HW, bias, res, out, M, N, K, act, out_half ? 1 : 0);
}

template <typename OT>
void layernorm(hipStream_t st, const float* x, LNW p, OT* out, int M, int C)
{
    k_layernorm<OT><<<ceil_div(M, 8), 256, 0, st>>>(x, p.g, p.b, out, M, C);
}

void swtb(hipStream_t st, const SwtbW& w, float* x, int Bt, int H, int Wd, int C,
          _Float16* LA, _Float16* HB, _Float16* HW)
{
    const int M = Bt * H * Wd;
    const int heads = C / 32;
    const int nh = H / 7, nw = Wd / 7;
    layernorm<_Float16>(st, x, w.ln1, LA, M, C);
    gemm(st, LA, w.qkv.w, w.qkv.b, nullptr, HB, true, M, 3 * C, C, 0, HW);
    k_win_attn<<<dim3(Bt * nh * nw * heads), dim3(32), 0, st>>>(HB, w.rpb, LA, H, Wd, C, heads, nh, nw);
    gemm(st, LA, w.proj.w, w.proj.b, x, x, false, M, C, C, 0, HW);       // residual in-place
    layernorm<_Float16>(st, x, w.ln2, LA, M, C);
    gemm(st, LA, w.fc1.w, w.fc1.b, nullptr, HB, true, M, 4 * C, C, 1, HW);  // GELU, f16 out
    gemm(st, HB, w.fc2.w, w.fc2.b, x, x, false, M, C, 4 * C, 0, HW);     // residual in-place
}

void swin_block(hipStream_t st, const BlockW& b, float* x, int Bt, int H, int Wd, int C,
                _Float16* LA, _Float16* HB, _Float16* HW)
{
    swtb(st, b.b1, x, Bt, H, Wd, C, LA, HB, HW);
    swtb(st, b.b2, x, Bt, H, Wd, C, LA, HB, HW);
}

#define EW(kern, n, ...) kern<<<ceil_div((n), 256), 256, 0, stream>>>(__VA_ARGS__)

}  // namespace

extern "C" void kernel_launch(void* const* d_in, const int* in_sizes, int n_in,
                              void* d_out, int out_size, void* d_ws, size_t ws_size,
                              hipStream_t stream)
{
    Reader rd{d_in, 0};
    const float* x = rd.f();                     // [4,3,448,448]
    LinW embed = rdLin(rd);                      // w (96,3,4,4), b (96)
    BlockW enc[3]; for (int i = 0; i < 3; ++i) enc[i] = rdBlock(rd);
    struct MergeW { LNW ln; const float* red; } mrg[3];
    for (int i = 0; i < 3; ++i) { mrg[i].ln = rdLN(rd); mrg[i].red = rd.f(); }
    BlockW bott = rdBlock(rd);
    struct ExpW { const float* exp; LNW ln; } expw[3];
    for (int i = 0; i < 3; ++i) { expw[i].exp = rd.f(); expw[i].ln = rdLN(rd); }
    LinW dskip[3]; for (int i = 0; i < 3; ++i) dskip[i] = rdLin(rd);
    BlockW dec[3]; for (int i = 0; i < 3; ++i) dec[i] = rdBlock(rd);
    const float* fin_w = rd.f(); const float* fin_b = rd.f(); LNW fin_ln = rdLN(rd);
    LinW head = rdLin(rd);

    // --- workspace arena (bytes) -------------------------------------------
    char* base = (char*)d_ws; size_t off = 0;
    auto alloc = [&](size_t bytes) -> void* {
        size_t a = (off + 255) & ~(size_t)255;
        off = a + bytes;
        return (void*)(base + a);
    };
    float*    T   = (float*)alloc(4816896ull * 4);   // tokens, max [50176,96] f32
    float*    SA  = (float*)alloc(9633792ull * 4);   // f32 scratch A
    float*    SB  = (float*)alloc(9633792ull * 4);   // f32 scratch B
    float*    s0  = (float*)alloc(4816896ull * 4);   // skip 112^2 x 96
    float*    s1  = (float*)alloc(2408448ull * 4);   // skip 56^2  x 192
    float*    s2  = (float*)alloc(1204224ull * 4);   // skip 28^2  x 384
    _Float16* HB  = (_Float16*)alloc(19267584ull * 2);  // f16 big (qkv/fc1/concat/cvt)
    _Float16* LA  = (_Float16*)alloc(9633792ull * 2);   // f16 (LN out / attn out)
    _Float16* HW  = (_Float16*)alloc(2359296ull * 2);   // f16 weight scratch [N,K]
    float* embedWT = (float*)alloc(48 * 96 * 4);
    float* finWT   = (float*)alloc(96 * 192 * 4);
    float* headWT  = (float*)alloc(48 * 24 * 4);
    float* headB   = (float*)alloc(24 * 4);
    if (off > ws_size) return;  // workspace too small: deterministic no-op

    const int Bt = 4;

    // --- weight repacks (to f32 [K,N]; gemm() transposes to f16 [N,K]) ------
    EW(k_transpose, 96 * 48, embed.w, embedWT, 96, 48);   // (96,48)->[48,96]
    EW(k_repack_final, 96 * 192, fin_w, finWT);           // (96,48,2,2)->[96,192]
    EW(k_head_repack, 48 * 24, head.w, headWT);           // (19,48)->[48,24] padded
    EW(k_head_bias, 24, head.b, headB);

    // --- patch embed: im2col(f16) + GEMM -----------------------------------
    EW(k_im2col, 4 * 112 * 112 * 48, x, HB);
    gemm(stream, HB, embedWT, embed.b, nullptr, T, false, 50176, 96, 48, 0, HW);

    // --- encoder ------------------------------------------------------------
    const int encH[3] = {112, 56, 28};
    const int encC[3] = {96, 192, 384};
    float* skips[3] = {s0, s1, s2};
    for (int i = 0; i < 3; ++i) {
        int H = encH[i], C = encC[i], M = Bt * H * H;
        swin_block(stream, enc[i], T, Bt, H, H, C, LA, HB, HW);
        (void)hipMemcpyAsync(skips[i], T, (size_t)M * C * sizeof(float),
                             hipMemcpyDeviceToDevice, stream);
        // patch merge: gather -> LN(4C) f16 -> GEMM red (4C -> 2C)
        EW(k_merge_gather, (M / 4) * 4 * C, T, SA, Bt, H, H, C);
        layernorm<_Float16>(stream, SA, mrg[i].ln, LA, M / 4, 4 * C);
        gemm(stream, LA, mrg[i].red, nullptr, nullptr, T, false, M / 4, 2 * C, 4 * C, 0, HW);
    }

    // --- bottleneck (14^2 x 768) -------------------------------------------
    swin_block(stream, bott, T, Bt, 14, 14, 768, LA, HB, HW);

    // --- decoder ------------------------------------------------------------
    const int decHin[3] = {14, 28, 56};
    const int decCin[3] = {768, 384, 192};
    for (int i = 0; i < 3; ++i) {
        int H = decHin[i], C = decCin[i], M = Bt * H * H;
        int H2 = 2 * H, Co = C / 2, M4 = 4 * M;
        // patch_expand: cvt A -> GEMM (C -> 2C) -> rearrange -> LN (f32, stays tokens)
        EW(k_cvt, M * C, T, HB, M * C);
        gemm(stream, HB, expw[i].exp, nullptr, nullptr, SA, false, M, 2 * C, C, 0, HW);
        EW(k_expand_rearrange, M4 * Co, SA, SB, Bt, H, H, Co);
        layernorm<float>(stream, SB, expw[i].ln, T, M4, Co);
        // concat with skip (f16 out) -> dec_skip GEMM (C -> C/2)
        EW(k_concat_h, M4 * C, T, skips[2 - i], HB, M4, Co, Co);
        gemm(stream, HB, dskip[i].w, dskip[i].b, nullptr, T, false, M4, Co, C, 0, HW);
        swin_block(stream, dec[i], T, Bt, H2, H2, Co, LA, HB, HW);
    }

    // --- final 2x2 transpose-conv + LN -------------------------------------
    EW(k_cvt, 50176 * 96, T, HB, 50176 * 96);
    gemm(stream, HB, finWT, nullptr, nullptr, SB, false, 50176, 192, 96, 0, HW);
    EW(k_final_rearrange, 4 * 224 * 224 * 48, SB, fin_b, SA);   // + bias
    layernorm<_Float16>(stream, SA, fin_ln, LA, 4 * 224 * 224, 48);

    // --- 1x1 head (N padded to 24) + bilinear resize to 480 -----------------
    gemm(stream, LA, headWT, headB, nullptr, SB, false, 4 * 224 * 224, 24, 48, 0, HW);
    EW(k_resize, 4 * 19 * 480 * 480, SB, (float*)d_out);
}
